// iso_v3_79594333929774
// MI455X (gfx1250) — matrix operand order repaired and checked by reading, not compiled, not run on hardware
//
#include <hip/hip_runtime.h>
#include <hip/hip_bf16.h>
#include <math.h>

typedef __attribute__((ext_vector_type(16))) _Float16 v16h;
typedef __attribute__((ext_vector_type(8)))  float    v8f;

#define NB    128
#define CDIM  164
#define TDIM  401
#define TPADL 32
#define TP    512   // padded time stride (halves)
#define CP    192   // padded ci (K)
#define COP   192   // padded co (M), 6 blocks of 32
#define NCOB  6     // co blocks of 32
#define NTB   7     // t blocks of 64 (448 >= 401)
#define GATES 656
#define HID   164
#define BATCH 16
#define TSEQ  496
#define KP0   192
#define KP1   352
#define HSLD  352

__device__ __forceinline__ float sigf(float x) { return 1.f / (1.f + expf(-x)); }

__device__ __forceinline__ v8f wmma16(v16h a, v16h b, v8f c) {
  return __builtin_amdgcn_wmma_f32_16x16x32_f16(false, a, false, b, (short)0, c, false, false);
}

// load A fragment: rows [row0 + (lane&15)], two contiguous 8-half chunks
__device__ __forceinline__ v16h load_afrag(const _Float16* __restrict__ p) {
  v16h a;
#pragma unroll
  for (int e = 0; e < 8; e++) a[e] = p[e];
#pragma unroll
  for (int e = 0; e < 8; e++) a[8 + e] = p[16 + e];
  return a;
}
// load B fragment: column n fixed, rows K=koff+e and K=16+koff+e (stride ld)
__device__ __forceinline__ v16h load_bfrag(const _Float16* __restrict__ p, int ld) {
  v16h b;
#pragma unroll
  for (int e = 0; e < 8; e++) b[e] = p[(size_t)e * ld];
#pragma unroll
  for (int e = 0; e < 8; e++) b[8 + e] = p[(size_t)(16 + e) * ld];
  return b;
}

// ---------- weight conversion ----------
// src f32 [164][164][11] -> dst f16 [11][192][192], zero padded
__global__ void k_cvt_convw(const float* __restrict__ src, _Float16* __restrict__ dst) {
  int i = blockIdx.x * blockDim.x + threadIdx.x;
  if (i >= 11 * COP * CP) return;
  int ci = i % CP; int r = i / CP; int co = r % COP; int tap = r / COP;
  float v = 0.f;
  if (co < CDIM && ci < CDIM) v = src[(co * CDIM + ci) * 11 + tap];
  dst[i] = (_Float16)v;
}

// W f32 [N][K] -> dst f16 [Kpad][N] (transposed, k>=K zero)
__global__ void k_cvt_wT(const float* __restrict__ W, _Float16* __restrict__ dst,
                         int N, int K, int Kpad) {
  int i = blockIdx.x * blockDim.x + threadIdx.x;
  if (i >= Kpad * N) return;
  int n = i % N; int k = i / N;
  dst[i] = (_Float16)((k < K) ? W[(size_t)n * K + k] : 0.f);
}

// ---------- simple FC: out[m][n] = act(b[n] + sum_k A[m][k]*W[n][k]) ----------
__global__ void k_fc(const float* __restrict__ A, const float* __restrict__ W,
                     const float* __restrict__ b, float* __restrict__ out,
                     int M, int N, int K, int relu) {
  int i = blockIdx.x * blockDim.x + threadIdx.x;
  if (i >= M * N) return;
  int m = i / N, n = i % N;
  const float* a = A + (size_t)m * K;
  const float* w = W + (size_t)n * K;
  float acc = b[n];
  for (int k = 0; k < K; k++) acc += a[k] * w[k];
  out[i] = relu ? fmaxf(acc, 0.f) : acc;
}

// ---------- pangolin front: conv1 4->164 (k=1) ----------
__global__ void k_conv1(const float* __restrict__ x, const float* __restrict__ w,
                        const float* __restrict__ b, float* __restrict__ out) {
  int i = blockIdx.x * blockDim.x + threadIdx.x;
  if (i >= NB * CDIM * TDIM) return;
  int t = i % TDIM; int r = i / TDIM; int c = r % CDIM; int n = r / CDIM;
  const float* xn = x + (size_t)n * 4 * TDIM + t;
  float acc = b[c];
#pragma unroll
  for (int j = 0; j < 4; j++) acc += w[c * 4 + j] * xn[(size_t)j * TDIM];
  out[i] = acc;
}

// ---------- BN + ReLU -> padded f16 activation buffer ----------
__global__ void k_bnrelu(const float* __restrict__ conv, const float* __restrict__ g,
                         const float* __restrict__ bt, const float* __restrict__ mn,
                         const float* __restrict__ vr, _Float16* __restrict__ act) {
  int i = blockIdx.x * blockDim.x + threadIdx.x;
  if (i >= NB * CDIM * TDIM) return;
  int t = i % TDIM; int r = i / TDIM; int c = r % CDIM; int n = r / CDIM;
  float sc = g[c] * rsqrtf(vr[c] + 1e-5f);
  float v = (conv[i] - mn[c]) * sc + bt[c];
  act[((size_t)n * CP + c) * TP + t + TPADL] = (_Float16)fmaxf(v, 0.f);
}

// ---------- WMMA implicit-GEMM dilated conv (k=11), 32co x 64t per wave ----------
// act f16 [NB][CP][TP] (zero-padded), Wt f16 [11][COP][CP]
__global__ void k_wmma_conv(const _Float16* __restrict__ act, const _Float16* __restrict__ Wt,
                            const float* __restrict__ bias, const float* resid,
                            float* out, int dil) {
  int wave = (blockIdx.x * blockDim.x + threadIdx.x) >> 5;
  int lane = threadIdx.x & 31;
  if (wave >= NB * NTB * NCOB) return;
  int cb = wave % NCOB;
  int tmp = wave / NCOB;
  int tb = tmp % NTB;
  int n = tmp / NTB;
  int co0 = cb * 32;
  int t0 = tb * 64;

  int m = lane & 15;
  int koff = (lane & 16) ? 8 : 0;
  v8f acc[8];  // acc[2*j+h]: t sub-tile j (0..3), co half h (0..1)
#pragma unroll
  for (int q = 0; q < 8; q++) acc[q] = (v8f){};

  const _Float16* actn = act + (size_t)n * CP * TP;
  for (int tap = 0; tap < 11; ++tap) {
    int shift = (tap - 5) * dil;
    const _Float16* wtap0 = Wt + ((size_t)tap * COP + (co0 + m)) * CP + koff;
    const _Float16* wtap1 = wtap0 + (size_t)16 * CP;
    const _Float16* btap = actn + t0 + shift + TPADL;
    for (int kc = 0; kc < 6; ++kc) {
      int kb = kc * 32;
      v16h a0 = load_afrag(wtap0 + kb);
      v16h a1 = load_afrag(wtap1 + kb);
      const _Float16* brow = btap + (size_t)(kb + koff) * TP + m;
      if (kc < 5) __builtin_prefetch(brow + 32 * TP, 0, 1);
#pragma unroll
      for (int j = 0; j < 4; ++j) {
        v16h bfr = load_bfrag(brow + j * 16, TP);
        acc[2 * j + 0] = wmma16(a0, bfr, acc[2 * j + 0]);
        acc[2 * j + 1] = wmma16(a1, bfr, acc[2 * j + 1]);
      }
    }
  }
  int rofs = (lane & 16) ? 8 : 0;
#pragma unroll
  for (int j = 0; j < 4; ++j) {
    int col = t0 + j * 16 + m;
    if (col >= TDIM) continue;
#pragma unroll
    for (int h = 0; h < 2; ++h) {
#pragma unroll
      for (int r = 0; r < 8; r++) {
        int co = co0 + h * 16 + r + rofs;
        if (co < CDIM) {
          size_t oi = ((size_t)n * CDIM + co) * TDIM + col;
          float v = acc[2 * j + h][r] + bias[co];
          if (resid) v += resid[oi];
          out[oi] = v;
        }
      }
    }
  }
}

// ---------- 1x1 conv evaluated only at t=200 (skip path) ----------
__global__ void k_gemm_t200(const float* __restrict__ conv, const float* __restrict__ W,
                            const float* __restrict__ b, float* out, int accum) {
  int i = blockIdx.x * blockDim.x + threadIdx.x;
  if (i >= NB * CDIM) return;
  int co = i % CDIM, n = i / CDIM;
  const float* src = conv + (size_t)n * CDIM * TDIM + 200;
  const float* w = W + (size_t)co * CDIM;
  float acc = b[co];
  for (int ci = 0; ci < CDIM; ci++) acc += w[ci] * src[(size_t)ci * TDIM];
  out[i] = accum ? out[i] + acc : acc;
}

// ---------- fc1/fc2: concat(r[sid], pang) @ W^T + b, relu ----------
__global__ void k_fc_concat(const float* __restrict__ r, const float* __restrict__ pg,
                            const int* __restrict__ lens, const float* __restrict__ W,
                            const float* __restrict__ b, float* __restrict__ out) {
  int i = blockIdx.x * blockDim.x + threadIdx.x;
  if (i >= NB * CDIM) return;
  int j = i % CDIM, n = i / CDIM;
  int sid = 0;
  for (int s = 1; s <= BATCH; s++)
    if (lens[s] <= n) sid = s;
  const float* w = W + (size_t)j * (2 * CDIM);
  const float* rr = r + (size_t)sid * CDIM;
  const float* pp = pg + (size_t)n * CDIM;
  float acc = b[j];
  for (int k = 0; k < CDIM; k++) acc += rr[k] * w[k];
  for (int k = 0; k < CDIM; k++) acc += pp[k] * w[CDIM + k];
  out[i] = fmaxf(acc, 0.f);
}

// ---------- gather pool rows * emb_table[label] -> f16 [t][b][192] ----------
__global__ void k_build_emb(const float* __restrict__ fc1o, const float* __restrict__ fc2o,
                            const float* __restrict__ ssf, const float* __restrict__ etab,
                            const int* __restrict__ labels, const int* __restrict__ tpt,
                            const int* __restrict__ tpi, const int* __restrict__ ltss,
                            const int* __restrict__ ltes, const int* __restrict__ lss,
                            _Float16* __restrict__ embtb) {
  int i = blockIdx.x * blockDim.x + threadIdx.x;
  if (i >= BATCH * TSEQ * KP0) return;
  int c = i % KP0; int bt = i / KP0; int b = bt % BATCH; int t = bt / BATCH;
  _Float16 v = (_Float16)0.f;
  if (c < CDIM) {
    int ty = tpt[b * TSEQ + t];
    int ix = tpi[b * TSEQ + t];
    const float* src;
    if (ty == 0)      src = fc1o + (size_t)(ltss[b] + ix) * CDIM;
    else if (ty == 1) src = fc2o + (size_t)(ltes[b] + ix) * CDIM;
    else              src = ssf + (size_t)(lss[b] + ix) * CDIM;
    int lab = labels[b * TSEQ + t];
    v = (_Float16)(src[c] * etab[lab * CDIM + c]);
  }
  embtb[((size_t)t * BATCH + b) * KP0 + c] = v;
}

// ---------- per-sequence row reversal (gather), f16, generic ld ----------
__global__ void k_reverse_rows(const _Float16* __restrict__ in, _Float16* __restrict__ out,
                               const int* __restrict__ seqlen, int ld) {
  int i = blockIdx.x * blockDim.x + threadIdx.x;
  if (i >= TSEQ * BATCH * ld) return;
  int c = i % ld; int bt = i / ld; int b = bt % BATCH; int t = bt / BATCH;
  int L = seqlen[b];
  int ts = (t < L) ? (L - 1 - t) : t;
  out[((size_t)t * BATCH + b) * ld + c] = in[((size_t)ts * BATCH + b) * ld + c];
}

// ---------- WMMA GEMM, 64 rows x 16 cols per wave (B frag reused 4x) ----------
// out = A(f16,[rows][Kpad]) x BT(f16,[Kpad][N]) + b1 + b2 ; mtiles multiple of 4
__global__ void k_wmma_gemm(const _Float16* __restrict__ A, const _Float16* __restrict__ BT,
                            const float* __restrict__ b1, const float* __restrict__ b2,
                            float* __restrict__ out, int mtiles, int ntiles, int kchunks,
                            int Kpad, int N) {
  int wave = (blockIdx.x * blockDim.x + threadIdx.x) >> 5;
  int lane = threadIdx.x & 31;
  int mblocks = mtiles >> 2;
  if (wave >= mblocks * ntiles) return;
  int nt = wave % ntiles, mb = wave / ntiles;
  int m = lane & 15;
  int koff = (lane & 16) ? 8 : 0;
  int rofs = koff;
  v8f acc[4];
#pragma unroll
  for (int q = 0; q < 4; q++) acc[q] = (v8f){};
  const _Float16* Arow = A + (size_t)(mb * 64 + m) * Kpad + koff;
  const _Float16* Bcol = BT + (size_t)koff * N + nt * 16 + m;
  for (int kc = 0; kc < kchunks; kc++) {
    int kb = kc * 32;
    v16h bfr = load_bfrag(Bcol + (size_t)kb * N, N);
#pragma unroll
    for (int q = 0; q < 4; q++) {
      v16h a = load_afrag(Arow + (size_t)(q * 16) * Kpad + kb);
      acc[q] = wmma16(a, bfr, acc[q]);
    }
  }
  int col = nt * 16 + m;
  float badd = b1[col] + b2[col];
#pragma unroll
  for (int q = 0; q < 4; q++) {
#pragma unroll
    for (int r = 0; r < 8; r++) {
      int row = mb * 64 + q * 16 + r + rofs;
      out[(size_t)row * N + col] = acc[q][r] + badd;
    }
  }
}

// ---------- persistent BiLSTM scan: 1 block per direction, Whh^T resident in LDS ----------
__global__ void k_lstm_scan(const float* __restrict__ xprojF, const float* __restrict__ xprojB,
                            const _Float16* __restrict__ whhTF, const _Float16* __restrict__ whhTB,
                            const int* __restrict__ seqlen, _Float16* hs_out,
                            float* __restrict__ cfin, int store_hs) {
  extern __shared__ char smem[];
  _Float16* sW = (_Float16*)smem;                 // [192][656]
  _Float16* sH = sW + KP0 * GATES;                // [16][192], pad cols zero
  float* sC = (float*)(sH + BATCH * KP0);         // [16][164]
  float* sG = sC + BATCH * HID;                   // [16][656]

  int dir = blockIdx.x;
  const float* xp = dir ? xprojB : xprojF;
  const _Float16* whhT = dir ? whhTB : whhTF;
  int tid = threadIdx.x;
  for (int i = tid; i < KP0 * GATES; i += blockDim.x) sW[i] = whhT[i];
  for (int i = tid; i < BATCH * KP0; i += blockDim.x) sH[i] = (_Float16)0.f;
  for (int i = tid; i < BATCH * HID; i += blockDim.x) sC[i] = 0.f;
  __syncthreads();

  int wave = tid >> 5, lane = tid & 31;
  int m = lane & 15;
  int koff = (lane & 16) ? 8 : 0;
  int rofs = koff;
  for (int t = 0; t < TSEQ; t++) {
    const float* xpt = xp + (size_t)t * BATCH * GATES;
    for (int nt = wave; nt < 41; nt += 16) {
      int n0 = nt * 16;
      v8f acc;
#pragma unroll
      for (int r = 0; r < 8; r++) acc[r] = xpt[(size_t)(r + rofs) * GATES + n0 + m];
      for (int kc = 0; kc < 6; kc++) {
        int kb = kc * 32;
        v16h a = load_afrag(sH + (size_t)m * KP0 + kb + koff);
        v16h bfr = load_bfrag(sW + (size_t)(kb + koff) * GATES + n0 + m, GATES);
        acc = wmma16(a, bfr, acc);
      }
#pragma unroll
      for (int r = 0; r < 8; r++) sG[(size_t)(r + rofs) * GATES + n0 + m] = acc[r];
    }
    __syncthreads();
    for (int i = tid; i < BATCH * HID; i += blockDim.x) {
      int b = i / HID, u = i - b * HID;
      int L = seqlen[b];
      if (t < L) {
        float gi = sG[b * GATES + u];
        float gf = sG[b * GATES + HID + u];
        float gg = sG[b * GATES + 2 * HID + u];
        float go = sG[b * GATES + 3 * HID + u];
        float cv = sigf(gf) * sC[i] + sigf(gi) * tanhf(gg);
        sC[i] = cv;
        sH[b * KP0 + u] = (_Float16)(sigf(go) * tanhf(cv));
      }
      if (store_hs) {
        int tw = t;
        if (dir) { int L2 = seqlen[b]; tw = (t < L2) ? (L2 - 1 - t) : t; }
        hs_out[((size_t)tw * BATCH + b) * HSLD + dir * HID + u] = sH[b * KP0 + u];
      }
    }
    __syncthreads();
  }
  for (int i = tid; i < BATCH * HID; i += blockDim.x)
    cfin[(size_t)dir * BATCH * HID + i] = sC[i];
}

// ---------- final iso FC ----------
__global__ void k_iso1(const float* __restrict__ cf, const float* __restrict__ W0,
                       const float* __restrict__ b0, float* __restrict__ hid) {
  int i = blockIdx.x * blockDim.x + threadIdx.x;
  if (i >= 16 * 64) return;
  int b = i / 64, j = i % 64;
  const float* w = W0 + (size_t)j * 328;
  float acc = b0[j];
  for (int k = 0; k < 164; k++) acc += cf[b * 164 + k] * w[k];
  for (int k = 0; k < 164; k++) acc += cf[16 * 164 + b * 164 + k] * w[164 + k];
  hid[i] = fmaxf(acc, 0.f);
}
__global__ void k_iso2(const float* __restrict__ hid, const float* __restrict__ W1,
                       const float* __restrict__ b1, float* __restrict__ out) {
  int i = blockIdx.x * blockDim.x + threadIdx.x;
  if (i >= 16) return;
  float acc = b1[0];
  for (int j = 0; j < 64; j++) acc += hid[i * 64 + j] * W1[j];
  out[i] = 1.f / (1.f + expf(-acc));
}

// ================= host =================
enum {
  IN_SEQ_TSS = 0, IN_SEQ_TES = 1, IN_SS = 2, IN_RBPS = 3,
  P_EMB = 4, P_FC1W = 5, P_FC1B = 6, P_FC2W = 7, P_FC2B = 8,
  P_ISOW0 = 9, P_ISOB0 = 10, P_ISOW1 = 11, P_ISOB1 = 12,
  P_L0B = 13, P_L0F = 17, P_L1B = 21, P_L1F = 25,
  P_PG1 = 29, P_PG2 = 135,
  P_RBPW0 = 241, P_RBPB0 = 242, P_RBPW1 = 243, P_RBPB1 = 244,
  IN_LEN_TSS = 245, IN_LEN_TES = 246, IN_LEN_SS = 247,
  IN_LABELS = 248, IN_TPTYPE = 249, IN_TPIDX = 250, IN_SEQLEN = 251
};
// pg block layout: +0 c1b, +1 c1w, +2 db0, +3 db1, +4 dw0, +5 dw1, +6 lb, +7 lw,
// +8+i*12: {b1,b2, bn1 g/bt/m/v, bn2 g/bt/m/v, w1, w2}, +104 skb, +105 skw

#define LA(kern, n, ...) kern<<<dim3((unsigned)(((n) + 255) / 256)), dim3(256), 0, stream>>>(__VA_ARGS__)

extern "C" void kernel_launch(void* const* d_in, const int* in_sizes, int n_in,
                              void* d_out, int out_size, void* d_ws, size_t ws_size,
                              hipStream_t stream) {
  auto F = [&](int i) { return (const float*)d_in[i]; };
  auto I = [&](int i) { return (const int*)d_in[i]; };
  char* WS = (char*)d_ws;
  size_t cur = 0;
  auto alloc = [&](size_t bytes) { size_t o = cur; cur = (cur + bytes + 255) & ~(size_t)255; return o; };

  const size_t CONVW_E = (size_t)11 * COP * CP;           // halves per conv weight
  size_t o_wtc = alloc(32 * CONVW_E * 2);
  size_t o_wih0f = alloc((size_t)KP0 * GATES * 2), o_wih0b = alloc((size_t)KP0 * GATES * 2);
  size_t o_whh0f = alloc((size_t)KP0 * GATES * 2), o_whh0b = alloc((size_t)KP0 * GATES * 2);
  size_t o_wih1f = alloc((size_t)KP1 * GATES * 2), o_wih1b = alloc((size_t)KP1 * GATES * 2);
  size_t o_whh1f = alloc((size_t)KP0 * GATES * 2), o_whh1b = alloc((size_t)KP0 * GATES * 2);
  size_t o_r512 = alloc(16 * 512 * 4);
  size_t o_rvec = alloc(16 * 164 * 4);
  size_t o_pang1 = alloc((size_t)NB * CDIM * 4), o_pang2 = alloc((size_t)NB * CDIM * 4);
  size_t o_skip = alloc((size_t)NB * CDIM * 4);
  size_t o_fc1o = alloc((size_t)NB * CDIM * 4), o_fc2o = alloc((size_t)NB * CDIM * 4);
  size_t o_cf0 = alloc(2 * 16 * 164 * 4), o_cf1 = alloc(2 * 16 * 164 * 4);
  size_t o_hid = alloc(16 * 64 * 4);
  size_t o_u = cur;
  const size_t SZ_CONV = (size_t)NB * CDIM * TDIM * 4;
  const size_t SZ_ACT = (size_t)NB * CP * TP * 2;
  const size_t SZ_EMB = (size_t)TSEQ * BATCH * KP0 * 2;
  const size_t SZ_XP = (size_t)TSEQ * BATCH * GATES * 4;
  const size_t SZ_HS = (size_t)TSEQ * BATCH * HSLD * 2;
  // pangolin view of the union
  size_t o_convA = o_u, o_convB = o_u + SZ_CONV, o_act = o_u + 2 * SZ_CONV;
  // lstm view of the union (live only after pangolins finish)
  size_t o_embtb = o_u, o_embtbr = o_u + SZ_EMB;
  size_t o_xpF = o_u + 2 * SZ_EMB, o_xpB = o_xpF + SZ_XP;
  size_t o_hs1 = o_xpB + SZ_XP, o_hs1r = o_hs1 + SZ_HS;
  size_t need = o_u + 2 * SZ_CONV + SZ_ACT;
  if (ws_size < need) return;

  _Float16* wtc = (_Float16*)(WS + o_wtc);
  float* convA = (float*)(WS + o_convA);
  float* convB = (float*)(WS + o_convB);
  _Float16* act = (_Float16*)(WS + o_act);
  float* r512 = (float*)(WS + o_r512);
  float* rvec = (float*)(WS + o_rvec);
  float* skip = (float*)(WS + o_skip);

  // ---- weight conversions ----
  for (int pg = 0; pg < 2; pg++) {
    int base = pg ? P_PG2 : P_PG1;
    for (int i = 0; i < 8; i++) {
      int rb = base + 8 + i * 12;
      LA(k_cvt_convw, 11 * COP * CP, F(rb + 10), wtc + (size_t)(pg * 16 + i * 2 + 0) * CONVW_E);
      LA(k_cvt_convw, 11 * COP * CP, F(rb + 11), wtc + (size_t)(pg * 16 + i * 2 + 1) * CONVW_E);
    }
  }
  LA(k_cvt_wT, KP0 * GATES, F(P_L0F + 0), (_Float16*)(WS + o_wih0f), GATES, 164, KP0);
  LA(k_cvt_wT, KP0 * GATES, F(P_L0B + 0), (_Float16*)(WS + o_wih0b), GATES, 164, KP0);
  LA(k_cvt_wT, KP0 * GATES, F(P_L0F + 1), (_Float16*)(WS + o_whh0f), GATES, 164, KP0);
  LA(k_cvt_wT, KP0 * GATES, F(P_L0B + 1), (_Float16*)(WS + o_whh0b), GATES, 164, KP0);
  LA(k_cvt_wT, KP1 * GATES, F(P_L1F + 0), (_Float16*)(WS + o_wih1f), GATES, 328, KP1);
  LA(k_cvt_wT, KP1 * GATES, F(P_L1B + 0), (_Float16*)(WS + o_wih1b), GATES, 328, KP1);
  LA(k_cvt_wT, KP0 * GATES, F(P_L1F + 1), (_Float16*)(WS + o_whh1f), GATES, 164, KP0);
  LA(k_cvt_wT, KP0 * GATES, F(P_L1B + 1), (_Float16*)(WS + o_whh1b), GATES, 164, KP0);

  // ---- rbp FC ----
  LA(k_fc, 16 * 512, F(IN_RBPS), F(P_RBPW0), F(P_RBPB0), r512, 16, 512, 1499, 1);
  LA(k_fc, 16 * 164, r512, F(P_RBPW1), F(P_RBPB1), rvec, 16, 164, 512, 1);

  // ---- pangolins ----
  hipMemsetAsync(act, 0, SZ_ACT, stream);
  int cwaves = NB * NTB * NCOB;
  dim3 cgrid((unsigned)((cwaves * 32 + 255) / 256)), cblk(256);
  auto pangolin = [&](int base, int pgi, const float* src, float* pout) {
    LA(k_conv1, NB * CDIM * TDIM, src, F(base + 1), F(base + 0), convA);
    LA(k_gemm_t200, NB * CDIM, convA, F(base + 105), F(base + 104), skip, 0);
    for (int i = 0; i < 8; i++) {
      int rb = base + 8 + i * 12;
      int dil = (i < 4) ? 1 : 4;
      LA(k_bnrelu, NB * CDIM * TDIM, convA, F(rb + 2), F(rb + 3), F(rb + 4), F(rb + 5), act);
      k_wmma_conv<<<cgrid, cblk, 0, stream>>>(act, wtc + (size_t)(pgi * 16 + i * 2 + 0) * CONVW_E,
                                              F(rb + 0), nullptr, convB, dil);
      LA(k_bnrelu, NB * CDIM * TDIM, convB, F(rb + 6), F(rb + 7), F(rb + 8), F(rb + 9), act);
      k_wmma_conv<<<cgrid, cblk, 0, stream>>>(act, wtc + (size_t)(pgi * 16 + i * 2 + 1) * CONVW_E,
                                              F(rb + 1), convA, convA, dil);
      if (i == 3) LA(k_gemm_t200, NB * CDIM, convA, F(base + 4), F(base + 2), skip, 1);
      if (i == 7) LA(k_gemm_t200, NB * CDIM, convA, F(base + 5), F(base + 3), skip, 1);
    }
    LA(k_fc, NB * CDIM, skip, F(base + 7), F(base + 6), pout, NB, CDIM, CDIM, 0);
  };
  pangolin(P_PG1, 0, F(IN_SEQ_TSS), (float*)(WS + o_pang1));
  pangolin(P_PG2, 1, F(IN_SEQ_TES), (float*)(WS + o_pang2));

  // ---- fc1 / fc2 ----
  LA(k_fc_concat, NB * CDIM, rvec, (float*)(WS + o_pang1), I(IN_LEN_TSS), F(P_FC1W), F(P_FC1B), (float*)(WS + o_fc1o));
  LA(k_fc_concat, NB * CDIM, rvec, (float*)(WS + o_pang2), I(IN_LEN_TES), F(P_FC2W), F(P_FC2B), (float*)(WS + o_fc2o));

  // ---- embedding gather (pangolin buffers dead; union switches to LSTM view) ----
  LA(k_build_emb, BATCH * TSEQ * KP0, (float*)(WS + o_fc1o), (float*)(WS + o_fc2o), F(IN_SS),
     F(P_EMB), I(IN_LABELS), I(IN_TPTYPE), I(IN_TPIDX), I(IN_LEN_TSS), I(IN_LEN_TES), I(IN_LEN_SS),
     (_Float16*)(WS + o_embtb));
  LA(k_reverse_rows, TSEQ * BATCH * KP0, (_Float16*)(WS + o_embtb), (_Float16*)(WS + o_embtbr),
     I(IN_SEQLEN), KP0);

  // ---- layer-0 input projections (WMMA GEMM, 496 Mtiles -> 124 blocks x 41 Ntiles) ----
  int gw = (TSEQ / 4) * 41;
  dim3 ggrid((unsigned)((gw * 32 + 255) / 256)), gblk(256);
  k_wmma_gemm<<<ggrid, gblk, 0, stream>>>((_Float16*)(WS + o_embtb), (_Float16*)(WS + o_wih0f),
                                          F(P_L0F + 2), F(P_L0F + 3), (float*)(WS + o_xpF),
                                          TSEQ, 41, 6, KP0, GATES);
  k_wmma_gemm<<<ggrid, gblk, 0, stream>>>((_Float16*)(WS + o_embtbr), (_Float16*)(WS + o_wih0b),
                                          F(P_L0B + 2), F(P_L0B + 3), (float*)(WS + o_xpB),
                                          TSEQ, 41, 6, KP0, GATES);

  // ---- layer-0 scan ----
  hipMemsetAsync(WS + o_hs1, 0, SZ_HS, stream);
  size_t SMEM = (size_t)(KP0 * GATES + BATCH * KP0) * sizeof(_Float16) +
                (size_t)(BATCH * HID + BATCH * GATES) * sizeof(float);
  k_lstm_scan<<<dim3(2), dim3(512), SMEM, stream>>>((float*)(WS + o_xpF), (float*)(WS + o_xpB),
                                                    (_Float16*)(WS + o_whh0f), (_Float16*)(WS + o_whh0b),
                                                    I(IN_SEQLEN), (_Float16*)(WS + o_hs1),
                                                    (float*)(WS + o_cf0), 1);

  // ---- layer-1 ----
  LA(k_reverse_rows, TSEQ * BATCH * HSLD, (_Float16*)(WS + o_hs1), (_Float16*)(WS + o_hs1r),
     I(IN_SEQLEN), HSLD);
  k_wmma_gemm<<<ggrid, gblk, 0, stream>>>((_Float16*)(WS + o_hs1), (_Float16*)(WS + o_wih1f),
                                          F(P_L1F + 2), F(P_L1F + 3), (float*)(WS + o_xpF),
                                          TSEQ, 41, 11, KP1, GATES);
  k_wmma_gemm<<<ggrid, gblk, 0, stream>>>((_Float16*)(WS + o_hs1r), (_Float16*)(WS + o_wih1b),
                                          F(P_L1B + 2), F(P_L1B + 3), (float*)(WS + o_xpB),
                                          TSEQ, 41, 11, KP1, GATES);
  k_lstm_scan<<<dim3(2), dim3(512), SMEM, stream>>>((float*)(WS + o_xpF), (float*)(WS + o_xpB),
                                                    (_Float16*)(WS + o_whh1f), (_Float16*)(WS + o_whh1b),
                                                    I(IN_SEQLEN), nullptr,
                                                    (float*)(WS + o_cf1), 0);

  // ---- final FC + sigmoid ----
  LA(k_iso1, 16 * 64, (float*)(WS + o_cf1), F(P_ISOW0), F(P_ISOB0), (float*)(WS + o_hid));
  LA(k_iso2, 16, (float*)(WS + o_hid), F(P_ISOW1), F(P_ISOB1), (float*)d_out);
}
